// SCAN_48017734369663
// MI455X (gfx1250) — compile-verified, matrix operand order checked by reading, not run on hardware
//
#include <hip/hip_runtime.h>

// ---------------------------------------------------------------------------
// SCAN / IMRAM t2i score on MI455X (gfx1250, wave32).
//   C=I=32, R=36, T=40, D=1024.
//   Heavy math (343.6 GFLOP) = gated-memory GEMMs -> v_wmma_f32_16x16x32_bf16,
//   double-buffered LDS with async global->LDS weight tiles (ASYNCcnt path).
//   Attention passes (~12 GFLOP) = VALU with LDS tiling.
//   Dead-code cut: the gated update after the last iteration is skipped.
// ---------------------------------------------------------------------------

#define NB   32      // images == captions
#define NR   36      // regions
#define NT   40      // words
#define ND   1024    // feature dim
#define NPAIR (NR*NT)
#define MROWS (NB*NB*NT)   // 40960 rows of the gate GEMM
#define KTOT  (2*ND)       // 2048
#define KSTEPS (KTOT/32)   // 64

typedef __attribute__((ext_vector_type(16))) __bf16 v16bf;
typedef __attribute__((ext_vector_type(8)))  __bf16 v8bf;
typedef __attribute__((ext_vector_type(8)))  float  v8f;

static __device__ __forceinline__ unsigned short f32_to_bf16(float f) {
    union { float f; unsigned u; } v; v.f = f;
    unsigned r = v.u + 0x7FFFu + ((v.u >> 16) & 1u);   // round-to-nearest-even
    return (unsigned short)(r >> 16);
}

// Load a 16x(32-k-slice) bf16 WMMA fragment from an LDS row of 32 bf16.
// ISA 16-bit A/B layout: lane L holds k = {kh..kh+7, 16+kh..16+kh+7}, kh = (L>>4)*8.
static __device__ __forceinline__ v16bf load_frag(const unsigned short* row, int lane) {
    int kh = (lane >> 4) * 8;
    const __bf16* p = reinterpret_cast<const __bf16*>(row);
    v8bf lo = *reinterpret_cast<const v8bf*>(p + kh);
    v8bf hi = *reinterpret_cast<const v8bf*>(p + 16 + kh);
    return __builtin_shufflevector(lo, hi, 0,1,2,3,4,5,6,7,8,9,10,11,12,13,14,15);
}

// CDNA5 async copy: 16 bytes global -> LDS, tracked by ASYNCcnt (no VGPR trip).
static __device__ __forceinline__ void async_ld_b128(unsigned lds_off, const void* gptr) {
    asm volatile("global_load_async_to_lds_b128 %0, %1, off"
                 :: "v"(lds_off), "v"((unsigned long long)(size_t)gptr)
                 : "memory");
}
static __device__ __forceinline__ void wait_async0() {
    asm volatile("s_wait_asynccnt 0x0" ::: "memory");
}

// ---------------------------------------------------------------------------
// Kernel 1: convert W_lin / W_gate (f32 [2048 k][1024 n]) to TRANSPOSED bf16
// [1024 n][2048 k] so B tiles are contiguous 16B segments for async b128 copy.
// ---------------------------------------------------------------------------
__global__ void convert_weights(const float* __restrict__ wl, const float* __restrict__ wg,
                                unsigned short* __restrict__ wlT, unsigned short* __restrict__ wgT) {
    const int n1 = KTOT * ND;                  // 2,097,152 per matrix
    int idx = blockIdx.x * blockDim.x + threadIdx.x;
    if (idx < n1) {
        int k = idx >> 10, n = idx & (ND - 1);           // coalesced read
        wlT[(size_t)n * KTOT + k] = f32_to_bf16(wl[idx]);
    } else if (idx < 2 * n1) {
        int j = idx - n1;
        int k = j >> 10, n = j & (ND - 1);
        wgT[(size_t)n * KTOT + k] = f32_to_bf16(wg[j]);
    }
}

// ---------------------------------------------------------------------------
// Kernel 2: fused func_attention_t2i + cosine-sim row score, one block per
// (caption c, image i) pair.  qmode: 0 -> query = cap_emb[c] (broadcast),
// 1 -> query = qsrc[(c,i)].  Optionally writes wc (needed only for pass 1).
// ---------------------------------------------------------------------------
__global__ __launch_bounds__(256)
void attn_kernel(const float* __restrict__ img,   // [NB][NR][ND]
                 const float* __restrict__ cap,   // [NB][NT][ND]
                 const float* __restrict__ qsrc,  // [NB*NB*NT][ND] or null
                 float* __restrict__ wc_out,      // [NB*NB*NT][ND] or null
                 float* __restrict__ score,       // [NB_i][NB_c]
                 int qmode, int write_wc, int accumulate) {
    __shared__ float ctx_s[NR * 128];
    __shared__ float qry_s[NT * 128];
    __shared__ float attn_s[NPAIR];    // [r][t]
    __shared__ float attnW_s[NPAIR];   // [t][r] softmax weights
    __shared__ float norms_s[NR];
    __shared__ float red_dot[NT], red_wsq[NT], red_csq[NT];

    const int tid = threadIdx.x;
    const int c = blockIdx.x >> 5;
    const int i = blockIdx.x & 31;
    const int pair = c * NB + i;

    // ---- Phase 1: attn[r][t] = sum_d ctx[i,r,d] * query[c,i,t,d], D-tiled ----
    float acc[6];
    #pragma unroll
    for (int j = 0; j < 6; ++j) acc[j] = 0.f;

    for (int d0 = 0; d0 < ND; d0 += 128) {
        #pragma unroll
        for (int j = 0; j < 18; ++j) {                     // 36*128 / 256
            int idx = tid + 256 * j;
            int r = idx >> 7, k = idx & 127;
            ctx_s[idx] = img[(i * NR + r) * ND + d0 + k];
        }
        #pragma unroll
        for (int j = 0; j < 20; ++j) {                     // 40*128 / 256
            int idx = tid + 256 * j;
            int t = idx >> 7, k = idx & 127;
            qry_s[idx] = qmode ? qsrc[((size_t)(pair * NT + t)) * ND + d0 + k]
                               : cap[(c * NT + t) * ND + d0 + k];
        }
        __syncthreads();
        #pragma unroll
        for (int j = 0; j < 6; ++j) {
            int p = tid + 256 * j;
            if (p < NPAIR) {
                int r = p / NT, t = p % NT;
                const float* cr = &ctx_s[r * 128];
                const float* qr = &qry_s[t * 128];
                float a = 0.f;
                #pragma unroll 8
                for (int k = 0; k < 128; ++k) a = fmaf(cr[k], qr[k], a);
                acc[j] += a;
            }
        }
        __syncthreads();
    }

    // ---- Phase 2: leaky relu, stash attn[r][t] ----
    #pragma unroll
    for (int j = 0; j < 6; ++j) {
        int p = tid + 256 * j;
        if (p < NPAIR) {
            float v = acc[j];
            attn_s[p] = (v > 0.f) ? v : 0.1f * v;
        }
    }
    __syncthreads();

    // ---- Phase 3: l2norm over t per region r ----
    if (tid < NR) {
        float s = 0.f;
        #pragma unroll
        for (int t = 0; t < NT; ++t) { float v = attn_s[tid * NT + t]; s = fmaf(v, v, s); }
        norms_s[tid] = sqrtf(s) + 1e-8f;
    }
    __syncthreads();

    // ---- Phase 4: temperature softmax over r per word t; zero reductions ----
    if (tid < NT) {
        float mx = -3.4e38f;
        float vals[NR];
        #pragma unroll
        for (int r = 0; r < NR; ++r) {
            float v = attn_s[r * NT + tid] / norms_s[r] * 9.0f;
            vals[r] = v;
            mx = fmaxf(mx, v);
        }
        float sum = 0.f;
        #pragma unroll
        for (int r = 0; r < NR; ++r) { vals[r] = expf(vals[r] - mx); sum += vals[r]; }
        float inv = 1.f / sum;
        #pragma unroll
        for (int r = 0; r < NR; ++r) attnW_s[tid * NR + r] = vals[r] * inv;
    }
    if (tid >= 64 && tid < 64 + NT) {
        red_dot[tid - 64] = 0.f; red_wsq[tid - 64] = 0.f; red_csq[tid - 64] = 0.f;
    }
    __syncthreads();

    // ---- Phase 5: wc = attnW @ ctx, fused cosine-sim partials ----
    for (int t = 0; t < NT; ++t) {
        const float* aw = &attnW_s[t * NR];
        float pd = 0.f, pw = 0.f, pc = 0.f;
        #pragma unroll
        for (int dd = 0; dd < 4; ++dd) {
            int d = tid + dd * 256;
            float a = 0.f;
            #pragma unroll
            for (int r = 0; r < NR; ++r) a = fmaf(aw[r], img[(i * NR + r) * ND + d], a);
            if (write_wc) wc_out[((size_t)(pair * NT + t)) * ND + d] = a;
            float cv = cap[(c * NT + t) * ND + d];
            pd = fmaf(cv, a, pd); pw = fmaf(a, a, pw); pc = fmaf(cv, cv, pc);
        }
        atomicAdd(&red_dot[t], pd);
        atomicAdd(&red_wsq[t], pw);
        atomicAdd(&red_csq[t], pc);
    }
    __syncthreads();

    if (tid == 0) {
        float s = 0.f;
        #pragma unroll
        for (int t = 0; t < NT; ++t) {
            float denom = fmaxf(sqrtf(red_csq[t]) * sqrtf(red_wsq[t]), 1e-8f);
            s += red_dot[t] / denom;
        }
        s *= (1.f / NT);
        int o = i * NB + c;                     // score is [image][caption]
        score[o] = accumulate ? (score[o] + s) : s;
    }
}

// ---------------------------------------------------------------------------
// Kernel 3: gated memory update via WMMA bf16, software-pipelined.
//   A[m,k] = k<1024 ? cap_emb[c,t,k] : wc[m,k-1024]   (m = (c,i,t), M=40960)
//   u = tanh(A@W_lin + b_lin); g = sigmoid(A@W_gate + b_gate)
//   query2[m,n] = cap[c,t,n]*g + u*(1-g)
// 128x64 block tile, 256 threads (8 waves, 4x2 wave grid, 32x32 per wave).
// Double-buffered LDS: A staged via VALU f32->bf16 convert; B tiles fetched
// with global_load_async_to_lds_b128 from pre-transposed bf16 weights.
// ---------------------------------------------------------------------------
__global__ __launch_bounds__(256)
void gated_gemm(const float* __restrict__ cap,
                const float* __restrict__ wc,
                const unsigned short* __restrict__ wlT,   // bf16 [1024 n][2048 k]
                const unsigned short* __restrict__ wgT,   // bf16 [1024 n][2048 k]
                const float* __restrict__ bl,
                const float* __restrict__ bg,
                float* __restrict__ q2) {
    __shared__ unsigned short As[2][128 * 32];   // [m][k]
    __shared__ unsigned short Bl[2][64 * 32];    // [n][k]
    __shared__ unsigned short Bg[2][64 * 32];    // [n][k]

    const int tid  = threadIdx.x;
    const int lane = tid & 31;
    const int wave = tid >> 5;
    const int wm = (wave >> 1) * 32;         // 0,32,64,96
    const int wn = (wave & 1) * 32;          // 0,32
    const int m0 = blockIdx.y * 128;         // 320 blocks -> 40960 rows exactly
    const int n0 = blockIdx.x * 64;          // 16 blocks  -> 1024 cols exactly

    v8f acc_l[2][2], acc_g[2][2];
    #pragma unroll
    for (int mi = 0; mi < 2; ++mi)
        #pragma unroll
        for (int ni = 0; ni < 2; ++ni) { acc_l[mi][ni] = (v8f)0.f; acc_g[mi][ni] = (v8f)0.f; }

    // Per-thread async-B coordinates: 64 rows x 4 x 16B segments = 256 xfers.
    const int brow = tid >> 2;
    const int bseg = (tid & 3) * 8;          // bf16 elements

    auto stageA = [&](int buf, int k0) {
        #pragma unroll
        for (int e = 0; e < 16; ++e) {       // 128*32 / 256
            int idx = tid + 256 * e;
            int m = idx >> 5, k = idx & 31;  // consecutive tids -> consecutive k
            int mg = m0 + m, gk = k0 + k;
            int ct = mg / (NB * NT);
            int rm = mg % (NB * NT);
            int t  = rm % NT;
            float v = (gk < ND) ? cap[(ct * NT + t) * ND + gk]
                                : wc[(size_t)mg * ND + (gk - ND)];
            As[buf][m * 32 + k] = f32_to_bf16(v);
        }
    };
    auto stageB = [&](int buf, int k0) {
        unsigned ldsl = (unsigned)(size_t)&Bl[buf][brow * 32 + bseg];
        unsigned ldsg = (unsigned)(size_t)&Bg[buf][brow * 32 + bseg];
        async_ld_b128(ldsl, wlT + (size_t)(n0 + brow) * KTOT + k0 + bseg);
        async_ld_b128(ldsg, wgT + (size_t)(n0 + brow) * KTOT + k0 + bseg);
    };
    auto compute = [&](int buf) {
        v16bf a[2], b_l[2], b_g[2];
        #pragma unroll
        for (int mi = 0; mi < 2; ++mi)
            a[mi] = load_frag(&As[buf][(wm + mi * 16 + (lane & 15)) * 32], lane);
        #pragma unroll
        for (int ni = 0; ni < 2; ++ni) {
            b_l[ni] = load_frag(&Bl[buf][(wn + ni * 16 + (lane & 15)) * 32], lane);
            b_g[ni] = load_frag(&Bg[buf][(wn + ni * 16 + (lane & 15)) * 32], lane);
        }
        #pragma unroll
        for (int mi = 0; mi < 2; ++mi)
            #pragma unroll
            for (int ni = 0; ni < 2; ++ni) {
                acc_l[mi][ni] = __builtin_amdgcn_wmma_f32_16x16x32_bf16(
                    false, a[mi], false, b_l[ni], (short)0, acc_l[mi][ni], false, false);
                acc_g[mi][ni] = __builtin_amdgcn_wmma_f32_16x16x32_bf16(
                    false, a[mi], false, b_g[ni], (short)0, acc_g[mi][ni], false, false);
            }
    };

    // Pipeline prologue
    stageA(0, 0);
    stageB(0, 0);
    wait_async0();
    __syncthreads();

    // Main loop, ping-pong buffers; stage k+1 while computing k.
    for (int kk = 0; kk < KSTEPS; ++kk) {
        int cur = kk & 1;
        if (kk + 1 < KSTEPS) {
            stageA(cur ^ 1, (kk + 1) * 32);
            stageB(cur ^ 1, (kk + 1) * 32);
        }
        compute(cur);
        wait_async0();       // our async transfers for buf cur^1 have landed
        __syncthreads();     // everyone's staging done; cur-buffer reads done
    }

    // Epilogue: C/D layout -> M = v + 8*(lane>>4), N = lane&15.
    #pragma unroll
    for (int mi = 0; mi < 2; ++mi)
        #pragma unroll
        for (int ni = 0; ni < 2; ++ni)
            #pragma unroll
            for (int v = 0; v < 8; ++v) {
                int mg = m0 + wm + mi * 16 + v + 8 * (lane >> 4);
                int ng = n0 + wn + ni * 16 + (lane & 15);
                float u = tanhf(acc_l[mi][ni][v] + bl[ng]);
                float g = 1.f / (1.f + expf(-(acc_g[mi][ni][v] + bg[ng])));
                int ct = mg / (NB * NT);
                int rm = mg % (NB * NT);
                int t  = rm % NT;
                float qv = cap[(ct * NT + t) * ND + ng];
                q2[(size_t)mg * ND + ng] = qv * g + u * (1.f - g);
            }
}

// ---------------------------------------------------------------------------
extern "C" void kernel_launch(void* const* d_in, const int* in_sizes, int n_in,
                              void* d_out, int out_size, void* d_ws, size_t ws_size,
                              hipStream_t stream) {
    const float* img = (const float*)d_in[0];   // [32,36,1024]
    const float* cap = (const float*)d_in[1];   // [32,40,1024]
    const float* Wl  = (const float*)d_in[2];   // [2048,1024]
    const float* blv = (const float*)d_in[3];   // [1024]
    const float* Wg  = (const float*)d_in[4];   // [2048,1024]
    const float* bgv = (const float*)d_in[5];   // [1024]
    float* out = (float*)d_out;                 // [32,32]

    char* ws = (char*)d_ws;
    const size_t WC_BYTES = (size_t)MROWS * ND * sizeof(float);      // 167,772,160
    float* wc = (float*)(ws);
    float* q2 = (float*)(ws + WC_BYTES);
    unsigned short* wlT = (unsigned short*)(ws + 2 * WC_BYTES);
    unsigned short* wgT = (unsigned short*)(ws + 2 * WC_BYTES + (size_t)KTOT * ND * 2);

    // weights -> transposed bf16 (L2-resident thereafter)
    convert_weights<<<(2 * KTOT * ND + 255) / 256, 256, 0, stream>>>(Wl, Wg, wlT, wgT);
    // iteration 1: attention with query = cap_emb broadcast; writes wc + score
    attn_kernel<<<NB * NB, 256, 0, stream>>>(img, cap, nullptr, wc, out, 0, 1, 0);
    // gated memory (only live update): query2 from [cap | wc]
    gated_gemm<<<dim3(ND / 64, MROWS / 128), 256, 0, stream>>>(cap, wc, wlT, wgT, blv, bgv, q2);
    // iteration 2: attention with query2; accumulate score (final gate is dead code)
    attn_kernel<<<NB * NB, 256, 0, stream>>>(img, cap, q2, nullptr, out, 1, 0, 1);
}
